// ContrastiveMSELoss_21251498180683
// MI455X (gfx1250) — matrix operand order repaired and checked by reading, not compiled, not run on hardware
//
#include <hip/hip_runtime.h>

typedef __attribute__((ext_vector_type(16))) _Float16 v16h;
typedef __attribute__((ext_vector_type(8)))  float    v8f;
typedef __attribute__((ext_vector_type(4)))  float    f4;

#define N_ROWS   8192
#define DIM      256
#define NCLS     40
#define KSPLIT   64
#define KCHUNK   128   // rows of X per workgroup in the WMMA kernel

// workspace layout (in floats)
#define WS_Y     0        // Y[48][256] padded class sums (rows 40..47 stay 0)
#define WS_SC    12288    // S_c[64]
#define WS_NC    12352    // n_c[64] (exact small integers in f32)
#define WS_STOT  12416    // S_total
#define WS_TOTAL 12544

__global__ void cmse_init(float* __restrict__ ws) {
    int i = blockIdx.x * blockDim.x + threadIdx.x;
    if (i < WS_TOTAL) ws[i] = 0.0f;
}

// One wave per row: sq_i = |x_i|^2, atomically fold into S_c, n_c, S_total.
__global__ void __launch_bounds__(256) cmse_rowstats(const float* __restrict__ X,
                                                     const int* __restrict__ cls,
                                                     float* __restrict__ ws) {
    int wave = threadIdx.x >> 5;            // 8 waves (wave32) per block
    int lane = threadIdx.x & 31;
    int row  = blockIdx.x * 8 + wave;
    const float* xr = X + (size_t)row * DIM + lane * 8;
    f4 a = *(const f4*)(xr);
    f4 b = *(const f4*)(xr + 4);
    float s = a.x*a.x + a.y*a.y + a.z*a.z + a.w*a.w
            + b.x*b.x + b.y*b.y + b.z*b.z + b.w*b.w;
    #pragma unroll
    for (int off = 16; off > 0; off >>= 1) s += __shfl_xor(s, off, 32);
    if (lane == 0) {
        int c = cls[row];
        atomicAdd(&ws[WS_SC + c], s);
        atomicAdd(&ws[WS_NC + c], 1.0f);
        atomicAdd(&ws[WS_STOT], s);
    }
}

// Y = H^T * X via v_wmma_f32_16x16x32_f16.
// grid.x = KSPLIT blocks; block = 512 threads = 16 waves; wave w owns columns [16w,16w+16).
// Each block covers K rows [blockIdx.x*KCHUNK, +KCHUNK), accumulating 3 M-tiles (48 class rows).
__global__ void __launch_bounds__(512) cmse_classsum(const float* __restrict__ X,
                                                     const int* __restrict__ cls,
                                                     float* __restrict__ ws) {
    __shared__ int scls[KCHUNK];
    const int tid   = threadIdx.x;
    const int kbase = blockIdx.x * KCHUNK;
    if (tid < KCHUNK) scls[tid] = cls[kbase + tid];
    __syncthreads();

    const int wave = tid >> 5;
    const int lane = tid & 31;
    const int half = lane >> 4;     // which 16-lane half
    const int r    = lane & 15;     // A-matrix row M within tile
    const int d0   = wave * 16;     // output columns for this wave

    v8f acc0 = {}, acc1 = {}, acc2 = {};

    for (int kk = 0; kk < KCHUNK; kk += 32) {
        // ---- B tile (32x16 f16): lane holds column N = d0+(lane&15), element j -> K' (same order as A) ----
        const float* xb = X + (size_t)(kbase + kk) * DIM + d0 + r;
        v16h b;
        #pragma unroll
        for (int j = 0; j < 16; ++j) {
            int kp = (j & 7) + ((j >> 3) << 4) + (half << 3);
            b[j] = (_Float16)xb[(size_t)kp * DIM];
        }
        // ---- A tiles (16x32 one-hot): element j -> K' per ISA 16-bit A layout ----
        v16h a0, a1, a2;
        #pragma unroll
        for (int j = 0; j < 16; ++j) {
            int kp = (j & 7) + ((j >> 3) << 4) + (half << 3);
            int c  = scls[kk + kp];
            a0[j] = (_Float16)((c == r     ) ? 1.0f : 0.0f);
            a1[j] = (_Float16)((c == r + 16) ? 1.0f : 0.0f);
            a2[j] = (_Float16)((c == r + 32) ? 1.0f : 0.0f);
        }
        acc0 = __builtin_amdgcn_wmma_f32_16x16x32_f16(false, a0, false, b, (short)0, acc0, false, false);
        acc1 = __builtin_amdgcn_wmma_f32_16x16x32_f16(false, a1, false, b, (short)0, acc1, false, false);
        acc2 = __builtin_amdgcn_wmma_f32_16x16x32_f16(false, a2, false, b, (short)0, acc2, false, false);
    }

    // C/D layout: lane<16 -> N=lane, VGPR i -> M=i ; lane>=16 -> N=lane-16, M=8+i
    const int n     = (lane & 15) + d0;
    const int mbase = half * 8;
    #pragma unroll
    for (int i = 0; i < 8; ++i) {
        int m = mbase + i;
        atomicAdd(&ws[WS_Y + (m +  0) * DIM + n], acc0[i]);
        atomicAdd(&ws[WS_Y + (m + 16) * DIM + n], acc1[i]);
        atomicAdd(&ws[WS_Y + (m + 32) * DIM + n], acc2[i]);
    }
}

// loss = 1 + [4*Sum_c n_c*S_c - 2N*S_total - 4*Sum_c |M_c|^2 + 2*|M_total|^2] / (N^2 * D)
__global__ void __launch_bounds__(256) cmse_finalize(const float* __restrict__ ws,
                                                     float* __restrict__ out) {
    __shared__ double red[256];
    int d = threadIdx.x;                 // one column per thread
    double colsum = 0.0, mc2 = 0.0;
    #pragma unroll 4
    for (int c = 0; c < 48; ++c) {
        double y = (double)ws[WS_Y + c * DIM + d];
        colsum += y;
        mc2    += y * y;
    }
    red[d] = 2.0 * colsum * colsum - 4.0 * mc2;
    __syncthreads();
    for (int s = 128; s > 0; s >>= 1) {
        if (d < s) red[d] += red[d + s];
        __syncthreads();
    }
    if (d == 0) {
        double acc = red[0];
        for (int c = 0; c < NCLS; ++c)
            acc += 4.0 * (double)ws[WS_NC + c] * (double)ws[WS_SC + c];
        acc -= 2.0 * (double)N_ROWS * (double)ws[WS_STOT];
        double loss = 1.0 + acc / ((double)N_ROWS * (double)N_ROWS * (double)DIM);
        out[0] = (float)loss;
    }
}

extern "C" void kernel_launch(void* const* d_in, const int* in_sizes, int n_in,
                              void* d_out, int out_size, void* d_ws, size_t ws_size,
                              hipStream_t stream) {
    const float* X   = (const float*)d_in[0];   // [8192, 256] f32
    const int*   cls = (const int*)d_in[1];     // [8192] int
    float*       out = (float*)d_out;           // scalar
    float*       ws  = (float*)d_ws;

    cmse_init<<<(WS_TOTAL + 255) / 256, 256, 0, stream>>>(ws);
    cmse_rowstats<<<N_ROWS / 8, 256, 0, stream>>>(X, cls, ws);
    cmse_classsum<<<KSPLIT, 512, 0, stream>>>(X, cls, ws);
    cmse_finalize<<<1, 256, 0, stream>>>(ws, out);
}
